// cola_gnn_40192303956392
// MI455X (gfx1250) — matrix operand order repaired and checked by reading, not compiled, not run on hardware
//
#include <hip/hip_runtime.h>
#include <hip/hip_bf16.h>
#include <math.h>

typedef __attribute__((ext_vector_type(16))) _Float16 v16h;
typedef __attribute__((ext_vector_type(8)))  float    v8f;

#define Bz 8
#define Mzz 512
#define Tz 20
#define Kc 13
#define Hh 64
#define HSPc 32
#define Pp 16
#define Nn (Bz*Mzz)        /* 4096 */
#define HALFc 32
#define LLc 8
#define RFEAT (Kc*(1+LLc)) /* 117 */

__device__ __forceinline__ float sigf(float x){ return 1.0f/(1.0f+__expf(-x)); }

// ---------------- LSTM: 16 nodes per 256-thread block ----------------
__global__ void lstm_kernel(const float* __restrict__ x,
                            const float* __restrict__ Wih,
                            const float* __restrict__ Whh,
                            const float* __restrict__ bih,
                            const float* __restrict__ bhh,
                            float* __restrict__ hlast)
{
  __shared__ float s_h[16][Hh];
  __shared__ float s_c[16][Hh];
  __shared__ float s_z[16][4*Hh];
  __shared__ float s_x[16][Tz];
  __shared__ float s_wih[4*Hh];
  __shared__ float s_b[4*Hh];
  int tid = threadIdx.x;
  int node0 = blockIdx.x*16;
  if (tid < 4*Hh) { s_wih[tid]=Wih[tid]; s_b[tid]=bih[tid]+bhh[tid]; }
  for (int i=tid; i<16*Tz; i+=256) {
    int ln = i/Tz, t = i%Tz;
    int n = node0+ln; int b = n>>9; int m = n & 511;
    s_x[ln][t] = x[(b*Tz + t)*Mzz + m];
  }
  for (int i=tid; i<16*Hh; i+=256) { (&s_h[0][0])[i]=0.f; (&s_c[0][0])[i]=0.f; }
  __syncthreads();
  int node = tid>>4;
  int g0 = (tid&15)*16;
  for (int t=0;t<Tz;t++){
    float xv = s_x[node][t];
    for (int gg=0; gg<16; gg++){
      int q = g0+gg;
      float acc = xv*s_wih[q] + s_b[q];
      const float* w = Whh + q*Hh;
      #pragma unroll 8
      for (int k=0;k<Hh;k++) acc += s_h[node][k]*w[k];
      s_z[node][q] = acc;
    }
    __syncthreads();
    for (int i=tid;i<16*Hh;i+=256){
      int ln=i>>6, hd=i&63;
      float zi=s_z[ln][hd], zf=s_z[ln][64+hd], zg=s_z[ln][128+hd], zo=s_z[ln][192+hd];
      float c = sigf(zf)*s_c[ln][hd] + sigf(zi)*tanhf(zg);
      s_c[ln][hd]=c;
      s_h[ln][hd]=sigf(zo)*tanhf(c);
    }
    __syncthreads();
  }
  for (int i=tid;i<16*Hh;i+=256){
    int ln=i>>6, hd=i&63;
    hlast[(node0+ln)*Hh+hd]=s_h[ln][hd];
  }
}

// ---------------- u = h@W2^T, v = h@W1^T ----------------
__global__ void uv_kernel(const float* __restrict__ h, const float* __restrict__ W1,
                          const float* __restrict__ W2, float* __restrict__ u, float* __restrict__ v)
{
  int idx = blockIdx.x*blockDim.x + threadIdx.x;
  if (idx >= Nn*64) return;
  int n = idx>>6; int j = idx&63;
  const float* hn = h + n*Hh;
  if (j<HALFc){
    const float* w = W2 + j*Hh; float acc=0.f;
    for (int k=0;k<Hh;k++) acc+=hn[k]*w[k];
    u[n*HALFc+j]=acc;
  } else {
    j-=HALFc;
    const float* w = W1 + j*Hh; float acc=0.f;
    for (int k=0;k<Hh;k++) acc+=hn[k]*w[k];
    v[n*HALFc+j]=acc;
  }
}

// ---------------- a[b,i,j] = elu(u_i + v_j + b1) @ V + bv ----------------
__global__ void attn_kernel(const float* __restrict__ u, const float* __restrict__ v,
                            const float* __restrict__ b1, const float* __restrict__ Vv,
                            const float* __restrict__ bv, float* __restrict__ a)
{
  __shared__ float su[HALFc], sb[HALFc], sV[HALFc];
  int b = blockIdx.z, i = blockIdx.y;
  int j = blockIdx.x*256 + threadIdx.x;
  if (threadIdx.x < HALFc){
    su[threadIdx.x]=u[(b*Mzz+i)*HALFc+threadIdx.x];
    sb[threadIdx.x]=b1[threadIdx.x];
    sV[threadIdx.x]=Vv[threadIdx.x];
  }
  __syncthreads();
  if (j>=Mzz) return;
  const float* vj = v + (b*Mzz+j)*HALFc;
  float acc = bv[0];
  #pragma unroll 8
  for (int k=0;k<HALFc;k++){
    float t = su[k]+vj[k]+sb[k];
    t = t>0.f ? t : (__expf(t)-1.0f);
    acc += t*sV[k];
  }
  a[((long)b*Mzz+i)*Mzz+j]=acc;
}

// ---------------- column L2 norms over i ----------------
__global__ void colnorm_kernel(const float* __restrict__ a, float* __restrict__ nrm)
{
  __shared__ float red[256];
  int bj = blockIdx.x; int b = bj>>9; int j = bj&511;
  const float* base = a + (long)b*Mzz*Mzz + j;
  float s=0.f;
  for (int i=threadIdx.x;i<Mzz;i+=256){ float t=base[(long)i*Mzz]; s+=t*t; }
  red[threadIdx.x]=s; __syncthreads();
  for (int st=128; st>0; st>>=1){
    if(threadIdx.x<st) red[threadIdx.x]+=red[threadIdx.x+st];
    __syncthreads();
  }
  if (threadIdx.x==0) nrm[bj]=sqrtf(red[0]);
}

__global__ void divnorm_kernel(float* __restrict__ a, const float* __restrict__ nrm, long n)
{
  long idx = (long)blockIdx.x*blockDim.x+threadIdx.x;
  if (idx>=n) return;
  int j = (int)(idx & 511);
  int b = (int)(idx >> 18);
  float d = nrm[b*Mzz+j];
  a[idx] /= fmaxf(d, 1e-12f);
}

// ---------------- f32 -> f16 convert (plain) ----------------
__global__ void cvt_f16_kernel(const float* __restrict__ s, _Float16* __restrict__ d, long n){
  long i=(long)blockIdx.x*blockDim.x+threadIdx.x;
  if(i<n) d[i]=(_Float16)s[i];
}

// ---------------- f32 -> f16 convert + transpose, batched ----------------
// src: (Kr x Nc) row-major ; dst: (Nc x Kr) row-major (i.e. B^T, N-major)
__global__ void cvt_f16_t_kernel(const float* __restrict__ s, _Float16* __restrict__ d,
                                 int Kr, int Nc, long sS, long sD)
{
  int idx = blockIdx.x*256 + threadIdx.x;
  if (idx >= Kr*Nc) return;
  int k = idx / Nc, n = idx % Nc;
  int bz = blockIdx.z;
  d[(long)bz*sD + (long)n*Kr + k] = (_Float16)s[(long)bz*sS + idx];
}

// ---------------- WMMA GEMM: C = A(MxK) @ B(KxN) (+bias,relu), batched ----------------
// B passed TRANSPOSED (BT: N x K row-major) so every fragment load is contiguous b128s.
// One wave per 32x32 output tile: 2 A-frags + 2 B-frags -> 4 WMMAs per k-step.
// blockDim.x == 32 (EXEC all ones, required by WMMA); K multiple of 32.
__global__ void wmma_gemm_kernel(const _Float16* __restrict__ A, const _Float16* __restrict__ BT,
                                 float* __restrict__ C, int Ndim, int Kdim,
                                 long sA, long sB, long sC,
                                 const float* __restrict__ bias, int relu)
{
  int lane = threadIdx.x & 31;
  int tn = blockIdx.x, tm = blockIdx.y, bz = blockIdx.z;
  int rc = lane & 15;      // A row / B,C column within 16-tile
  int hs = lane >> 4;      // half-wave select
  const _Float16* A0 = A  + (long)bz*sA + (long)(tm*32 + rc)*Kdim + hs*16;
  const _Float16* A1 = A0 + (long)16*Kdim;
  const _Float16* B0 = BT + (long)bz*sB + (long)(tn*32 + rc)*Kdim + hs*16;
  const _Float16* B1 = B0 + (long)16*Kdim;
  v8f acc00 = {}, acc01 = {}, acc10 = {}, acc11 = {};
  for (int k0=0;k0<Kdim;k0+=32){
    v16h a0,a1,b0,b1;
    #pragma unroll
    for (int jj=0;jj<8;jj++){
      a0[jj]=A0[k0+jj]; a0[8+jj]=A0[k0+8+jj];
      a1[jj]=A1[k0+jj]; a1[8+jj]=A1[k0+8+jj];
    }
    #pragma unroll
    for (int jj=0;jj<16;jj++){ b0[jj]=B0[k0+jj]; b1[jj]=B1[k0+jj]; }
    acc00 = __builtin_amdgcn_wmma_f32_16x16x32_f16(false,a0,false,b0,(short)0,acc00,false,false);
    acc01 = __builtin_amdgcn_wmma_f32_16x16x32_f16(false,a0,false,b1,(short)0,acc01,false,false);
    acc10 = __builtin_amdgcn_wmma_f32_16x16x32_f16(false,a1,false,b0,(short)0,acc10,false,false);
    acc11 = __builtin_amdgcn_wmma_f32_16x16x32_f16(false,a1,false,b1,(short)0,acc11,false,false);
  }
  float* Cb = C + (long)bz*sC + (long)(tm*32)*Ndim + tn*32;
  int rb2 = hs*8;
  #pragma unroll
  for (int ti=0; ti<2; ti++){
    #pragma unroll
    for (int tj=0; tj<2; tj++){
      v8f acc = (ti==0) ? (tj==0?acc00:acc01) : (tj==0?acc10:acc11);
      int colg = tn*32 + tj*16 + rc;
      float bia = bias ? bias[colg] : 0.f;
      #pragma unroll
      for (int vv=0; vv<8; vv++){
        float xv = acc[vv] + bia;
        if (relu) xv = fmaxf(xv, 0.f);
        Cb[(long)(ti*16 + rb2 + vv)*Ndim + tj*16 + rc] = xv;
      }
    }
  }
}

// ---------------- gated mix: a = adj*sig(t+wb) + a*(1-sig) ----------------
__global__ void gatemix_kernel(float* __restrict__ a, const float* __restrict__ t,
                               const float* __restrict__ adj, const float* __restrict__ wb, long n)
{
  long idx=(long)blockIdx.x*blockDim.x+threadIdx.x;
  if(idx>=n) return;
  int i = (int)((idx>>9)&511); int j = (int)(idx&511);
  float g = sigf(t[idx]+wb[0]);
  a[idx] = adj[i*Mzz+j]*g + a[idx]*(1.f-g);
}

// ---------------- per-node convs -> r (B,M,117) with relu ----------------
__global__ void conv_kernel(const float* __restrict__ x, const float* __restrict__ cw,
                            const float* __restrict__ cb, const float* __restrict__ clw,
                            const float* __restrict__ clb, float* __restrict__ r)
{
  int idx = blockIdx.x*256+threadIdx.x;
  if (idx>=Nn*Kc) return;
  int n = idx/Kc; int k = idx%Kc;
  int b=n>>9, m=n&511;
  float xs[Tz];
  #pragma unroll
  for(int t=0;t<Tz;t++) xs[t]=x[(b*Tz+t)*Mzz+m];
  float s = cb[k];
  const float* w = cw + k*Tz;
  #pragma unroll
  for(int t=0;t<Tz;t++) s += xs[t]*w[t];
  float* ro = r + (long)n*RFEAT + k*(1+LLc);
  ro[0]=fmaxf(s,0.f);
  const float* wl = clw + k*(Tz-Kc);
  for(int l=0;l<LLc;l++){
    float s2=clb[k];
    #pragma unroll
    for(int jj=0;jj<Tz-Kc;jj++) s2 += xs[l+2*jj]*wl[jj];
    ro[1+l]=fmaxf(s2,0.f);
  }
}

// ---------------- rg = r @ gc1_w (K=117 odd -> VALU) ----------------
__global__ void rg_kernel(const float* __restrict__ r, const float* __restrict__ w, float* __restrict__ rg)
{
  int idx=blockIdx.x*256+threadIdx.x; if(idx>=Nn*Hh) return;
  int n=idx>>6, o=idx&63;
  const float* rn = r + (long)n*RFEAT;
  float acc=0.f;
  for(int k=0;k<RFEAT;k++) acc += rn[k]*w[k*Hh+o];
  rg[idx]=acc;
}

// ---------------- final: concat-proj + autoregressive residual ----------------
__global__ void final_kernel(const float* __restrict__ spat, const float* __restrict__ hlast,
                             const float* __restrict__ x, const float* __restrict__ ow,
                             const float* __restrict__ ob, const float* __restrict__ rw,
                             const float* __restrict__ rb, float* __restrict__ out)
{
  int idx=blockIdx.x*256+threadIdx.x; if(idx>=Nn*Pp) return;
  int n=idx>>4, p=idx&15;
  int b=n>>9, m=n&511;
  float acc=ob[p]+rb[p];
  const float* sp = spat+(long)n*HSPc;
  #pragma unroll 8
  for(int k=0;k<HSPc;k++) acc+=sp[k]*ow[k*Pp+p];
  const float* hn = hlast+(long)n*Hh;
  #pragma unroll 8
  for(int k=0;k<Hh;k++) acc+=hn[k]*ow[(HSPc+k)*Pp+p];
  #pragma unroll
  for(int q=0;q<4;q++) acc += x[(b*Tz + (Tz-4+q))*Mzz + m]*rw[q*Pp+p];
  out[idx]=acc;
}

extern "C" void kernel_launch(void* const* d_in, const int* in_sizes, int n_in,
                              void* d_out, int out_size, void* d_ws, size_t ws_size,
                              hipStream_t stream)
{
  (void)in_sizes; (void)n_in; (void)out_size; (void)ws_size;
  const float* x      = (const float*)d_in[0];
  const float* adj    = (const float*)d_in[1];
  const float* W_ih   = (const float*)d_in[2];
  const float* W_hh   = (const float*)d_in[3];
  const float* b_ih   = (const float*)d_in[4];
  const float* b_hh   = (const float*)d_in[5];
  const float* W1     = (const float*)d_in[6];
  const float* W2     = (const float*)d_in[7];
  const float* b1     = (const float*)d_in[8];
  const float* Vv     = (const float*)d_in[9];
  const float* bv     = (const float*)d_in[10];
  const float* Wb     = (const float*)d_in[11];
  const float* wb     = (const float*)d_in[12];
  const float* conv_w = (const float*)d_in[13];
  const float* conv_b = (const float*)d_in[14];
  const float* convl_w= (const float*)d_in[15];
  const float* convl_b= (const float*)d_in[16];
  const float* gc1_w  = (const float*)d_in[17];
  const float* gc1_b  = (const float*)d_in[18];
  const float* gc2_w  = (const float*)d_in[19];
  const float* gc2_b  = (const float*)d_in[20];
  const float* out_w  = (const float*)d_in[21];
  const float* out_b  = (const float*)d_in[22];
  const float* res_w  = (const float*)d_in[23];
  const float* res_b  = (const float*)d_in[24];
  float* out = (float*)d_out;

  char* wsb = (char*)d_ws;
  size_t off = 0;
  auto alloc = [&](size_t bytes)->char*{
    char* p = wsb + off; off = (off + bytes + 255) & ~(size_t)255; return p;
  };
  const long AMM = (long)Bz*Mzz*Mzz;                 // 2,097,152
  float*    hlast  = (float*)   alloc((size_t)Nn*Hh*4);
  float*    ubuf   = (float*)   alloc((size_t)Nn*HALFc*4);
  float*    vbuf   = (float*)   alloc((size_t)Nn*HALFc*4);
  float*    abuf   = (float*)   alloc((size_t)AMM*4);
  float*    nrm    = (float*)   alloc((size_t)Bz*Mzz*4);
  float*    tbuf   = (float*)   alloc((size_t)AMM*4);
  _Float16* af16   = (_Float16*)alloc((size_t)AMM*2);
  _Float16* wbT    = (_Float16*)alloc((size_t)Mzz*Mzz*2);
  float*    rbuf   = (float*)   alloc((size_t)Nn*RFEAT*4);
  float*    rg     = (float*)   alloc((size_t)Nn*Hh*4);
  _Float16* rgT    = (_Float16*)alloc((size_t)Nn*Hh*2);
  float*    h1     = (float*)   alloc((size_t)Nn*Hh*4);
  _Float16* h1f16  = (_Float16*)alloc((size_t)Nn*Hh*2);
  _Float16* g2wT   = (_Float16*)alloc((size_t)Hh*HSPc*2);
  float*    h1g    = (float*)   alloc((size_t)Nn*HSPc*4);
  _Float16* h1gT   = (_Float16*)alloc((size_t)Nn*HSPc*2);
  float*    spat   = (float*)   alloc((size_t)Nn*HSPc*4);

  // 1. LSTM -> hlast (= out_temporal)
  lstm_kernel<<<Nn/16, 256, 0, stream>>>(x, W_ih, W_hh, b_ih, b_hh, hlast);
  // 2. u, v projections
  uv_kernel<<<(Nn*64)/256, 256, 0, stream>>>(hlast, W1, W2, ubuf, vbuf);
  // 3. attention scores a
  attn_kernel<<<dim3(2, Mzz, Bz), 256, 0, stream>>>(ubuf, vbuf, b1, Vv, bv, abuf);
  // 4-5. column L2 normalize
  colnorm_kernel<<<Bz*Mzz, 256, 0, stream>>>(abuf, nrm);
  divnorm_kernel<<<(int)((AMM+255)/256), 256, 0, stream>>>(abuf, nrm, AMM);
  // 6. convert a (plain) and Wb (transposed) to f16
  cvt_f16_kernel<<<(int)((AMM+255)/256), 256, 0, stream>>>(abuf, af16, AMM);
  cvt_f16_t_kernel<<<dim3((Mzz*Mzz)/256,1,1), 256, 0, stream>>>(Wb, wbT, Mzz, Mzz, 0L, 0L);
  // 7. t = a @ Wb   (8 x 512x512x512, WMMA, 32x32 tiles)
  wmma_gemm_kernel<<<dim3(Mzz/32, Mzz/32, Bz), 32, 0, stream>>>(
      af16, wbT, tbuf, Mzz, Mzz, (long)Mzz*Mzz, 0L, (long)Mzz*Mzz, nullptr, 0);
  // 8. gated mix in-place on a
  gatemix_kernel<<<(int)((AMM+255)/256), 256, 0, stream>>>(abuf, tbuf, adj, wb, AMM);
  // 9. conv features r
  conv_kernel<<<(Nn*Kc+255)/256, 256, 0, stream>>>(x, conv_w, conv_b, convl_w, convl_b, rbuf);
  // 10. rg = r @ gc1_w
  rg_kernel<<<(Nn*Hh)/256, 256, 0, stream>>>(rbuf, gc1_w, rg);
  // 11. reconvert gated a (plain); rg transposed per batch
  cvt_f16_kernel<<<(int)((AMM+255)/256), 256, 0, stream>>>(abuf, af16, AMM);
  cvt_f16_t_kernel<<<dim3((Mzz*Hh)/256,1,Bz), 256, 0, stream>>>(
      rg, rgT, Mzz, Hh, (long)Mzz*Hh, (long)Mzz*Hh);
  // 12. h1 = relu(a @ rg + gc1_b)   (8 x 512x512x64, WMMA)
  wmma_gemm_kernel<<<dim3(Hh/32, Mzz/32, Bz), 32, 0, stream>>>(
      af16, rgT, h1, Hh, Mzz, (long)Mzz*Mzz, (long)Mzz*Hh, (long)Mzz*Hh, gc1_b, 1);
  // 13. convert h1 (plain, it's the A operand), gc2_w (transposed)
  cvt_f16_kernel<<<(Nn*Hh)/256, 256, 0, stream>>>(h1, h1f16, (long)Nn*Hh);
  cvt_f16_t_kernel<<<dim3((Hh*HSPc+255)/256,1,1), 256, 0, stream>>>(
      gc2_w, g2wT, Hh, HSPc, 0L, 0L);
  // 14. h1g = h1 @ gc2_w   (single 4096x64x32, WMMA)
  wmma_gemm_kernel<<<dim3(HSPc/32, Nn/32, 1), 32, 0, stream>>>(
      h1f16, g2wT, h1g, HSPc, Hh, 0L, 0L, 0L, nullptr, 0);
  // 15. h1g transposed per batch (it's the B operand of the next GEMM)
  cvt_f16_t_kernel<<<dim3((Mzz*HSPc)/256,1,Bz), 256, 0, stream>>>(
      h1g, h1gT, Mzz, HSPc, (long)Mzz*HSPc, (long)Mzz*HSPc);
  // 16. out_spatial = relu(a @ h1g + gc2_b)   (8 x 512x512x32, WMMA)
  wmma_gemm_kernel<<<dim3(HSPc/32, Mzz/32, Bz), 32, 0, stream>>>(
      af16, h1gT, spat, HSPc, Mzz, (long)Mzz*Mzz, (long)Mzz*HSPc, (long)Mzz*HSPc, gc2_b, 1);
  // 17. final projection + autoregressive residual -> d_out
  final_kernel<<<(Nn*Pp)/256, 256, 0, stream>>>(spat, hlast, x, out_w, out_b, res_w, res_b, out);
}
